// MultiHeadAttention_69423851373089
// MI455X (gfx1250) — matrix operand order repaired and checked by reading, not compiled, not run on hardware
//
#include <hip/hip_runtime.h>
#include <hip/hip_bf16.h>

typedef __attribute__((ext_vector_type(16))) __bf16 v16bf;
typedef __attribute__((ext_vector_type(8)))  float  v8f;
typedef __attribute__((ext_vector_type(8)))  unsigned int u32x8;

#define EMB   256
#define NHEAD 8
#define HDIM  32

#if defined(__HIP_DEVICE_COMPILE__) && __has_builtin(__builtin_amdgcn_sched_barrier)
#define SCHED_FENCE() __builtin_amdgcn_sched_barrier(0)
#else
#define SCHED_FENCE() ((void)0)
#endif

// ---------- helpers ----------
__device__ __forceinline__ unsigned short f2bf(float f) {
  unsigned int x = __builtin_bit_cast(unsigned int, f);
  unsigned int r = x + 0x7FFFu + ((x >> 16) & 1u);   // round-to-nearest-even
  return (unsigned short)(r >> 16);
}
__device__ __forceinline__ float bflo(unsigned int u) { return __builtin_bit_cast(float, u << 16); }
__device__ __forceinline__ float bfhi(unsigned int u) { return __builtin_bit_cast(float, u & 0xFFFF0000u); }

__device__ __forceinline__ v16bf make_frag(uint4 a, uint4 b) {
  u32x8 u = {a.x, a.y, a.z, a.w, b.x, b.y, b.z, b.w};
  return __builtin_bit_cast(v16bf, u);
}

__device__ __forceinline__ void atomAddF(float* p, float v) {
  unsafeAtomicAdd(p, v);   // global_atomic_add_f32
}

// ---------- f32 -> bf16 conversion ----------
__global__ void cvt_bf16_kernel(const float* __restrict__ in,
                                unsigned short* __restrict__ out, int n) {
  int i = blockIdx.x * blockDim.x + threadIdx.x;
  if (i < n) out[i] = f2bf(in[i]);
}

__global__ void zero_kernel(float* __restrict__ p, int n) {
  int i = blockIdx.x * blockDim.x + threadIdx.x;
  if (i < n) p[i] = 0.0f;
}

// ---------- NT GEMM: C[M,256] = A[M,256](bf16) @ W[256,256](bf16, row-major [n,k])^T + bias ----------
// block = 128 threads = 4 waves; block covers a 16(M) x 256(N) tile; wave w owns N cols [64w, 64w+64).
// Per k-step: 1 A fragment + all 4 B fragments loaded (fenced), then 4 back-to-back WMMAs.
// The sched fence keeps loads batched and lets iteration k+1's loads pipeline into
// iteration k's WMMA phase.
__global__ __launch_bounds__(128) void gemm_nt_bf16_kernel(
    const unsigned short* __restrict__ A, const unsigned short* __restrict__ W,
    const float* __restrict__ bias, float* __restrict__ outF,
    unsigned short* __restrict__ outB, int M) {
  const int lane = threadIdx.x & 31;
  const int wave = threadIdx.x >> 5;
  const int m0   = blockIdx.x * 16;
  const int half = lane >> 4;
  const int l16  = lane & 15;

  // A-fragment row for this lane (clamped for tail safety; M is a multiple of 16 here)
  const unsigned short* arow = A + (size_t)min(m0 + l16, M - 1) * EMB;

  // Weight-row base pointer for this lane's column; the 4 N-tiles are +16 columns apart
  // (= +16*EMB elements), expressed as immediate offsets below.
  const unsigned short* wrow = W + (size_t)(wave * 64 + l16) * EMB;

  v8f acc[4] = {};

  #pragma unroll
  for (int kb = 0; kb < EMB; kb += 32) {
    // A fragment as loaded here (16x32): lane half -> K base 16*half; VGPR0-7 = K contiguous 16
    // (same K order as B)  => two B128 loads.
    const int kA = kb + 16 * half;
    uint4 alo = *(const uint4*)(arow + kA);
    uint4 ahi = *(const uint4*)(arow + kA + 8);

    // B fragment as loaded here (32x16): lane = column n, VGPR v holds K = 16*half + {2v,2v+1}
    // => contiguous 16-element K run from W row n => two B128 loads per tile.
    const int kB = kb + 16 * half;
    uint4 blo[4], bhi[4];
    #pragma unroll
    for (int nt = 0; nt < 4; ++nt) {
      blo[nt] = *(const uint4*)(wrow + (size_t)nt * 16 * EMB + kB);
      bhi[nt] = *(const uint4*)(wrow + (size_t)nt * 16 * EMB + kB + 8);
    }

    SCHED_FENCE();   // keep the 10 loads above, the 4 WMMAs below

    v16bf af = make_frag(alo, ahi);
    #pragma unroll
    for (int nt = 0; nt < 4; ++nt) {
      v16bf bfr = make_frag(blo[nt], bhi[nt]);
      acc[nt] = __builtin_amdgcn_wmma_f32_16x16x32_bf16(
          false, af, false, bfr, (short)0, acc[nt], false, false);
    }
  }

  // C/D layout: lane = column n (l16), half selects M+8; VGPR v -> row M = v + 8*half
  #pragma unroll
  for (int nt = 0; nt < 4; ++nt) {
    int col = wave * 64 + nt * 16 + l16;
    float bcol = bias ? bias[col] : 0.0f;
    #pragma unroll
    for (int v = 0; v < 8; ++v) {
      int row = m0 + v + 8 * half;
      if (row < M) {
        float val = acc[nt][v] + bcol;
        if (outF) outF[(size_t)row * EMB + col] = val;
        if (outB) outB[(size_t)row * EMB + col] = f2bf(val);
      }
    }
  }
}

// ---------- per-edge cross-head scores + segment-sum atomics ----------
// block = 256 threads = 8 waves; one edge per wave. qh/kh rows staged in LDS (512B each).
// lane computes scores s0 = lane (h=lane>>3, g=lane&7) and s1 = lane+32 (h+4, g).
__global__ __launch_bounds__(256) void edge_kernel(
    const unsigned short* __restrict__ qh, const unsigned short* __restrict__ kh,
    const int* __restrict__ ei, float* __restrict__ sums,
    float* __restrict__ counts, int E) {
  __shared__ unsigned short sq[8][EMB];
  __shared__ unsigned short sk[8][EMB];
  const int lane = threadIdx.x & 31;
  const int wave = threadIdx.x >> 5;
  int e = blockIdx.x * 8 + wave;
  bool valid = e < E;
  int ec = valid ? e : (E - 1);
  int a = ei[ec];
  int b = ei[E + ec];

  // cooperative row load: 32 lanes x 16B = 512B = full bf16 row
  ((uint4*)sq[wave])[lane] = ((const uint4*)(qh + (size_t)a * EMB))[lane];
  ((uint4*)sk[wave])[lane] = ((const uint4*)(kh + (size_t)b * EMB))[lane];
  __syncthreads();

  const int h0 = lane >> 3;   // 0..3
  const int g  = lane & 7;    // 0..7
  const unsigned int* q0 = (const unsigned int*)(sq[wave] + h0 * HDIM);
  const unsigned int* q1 = (const unsigned int*)(sq[wave] + (h0 + 4) * HDIM);
  const unsigned int* kk = (const unsigned int*)(sk[wave] + g * HDIM);

  float acc0 = 0.f, acc1 = 0.f;
  #pragma unroll
  for (int i = 0; i < 16; ++i) {   // 16 dwords = 32 bf16 per row
    unsigned int ku = kk[i];
    float k0 = bflo(ku), k1 = bfhi(ku);
    unsigned int qa = q0[i], qb = q1[i];
    acc0 += bflo(qa) * k0 + bfhi(qa) * k1;
    acc1 += bflo(qb) * k0 + bfhi(qb) * k1;
  }

  if (valid) {
    float* base = sums + (size_t)a * 64;
    atomAddF(base + h0 * 8 + g, acc0);
    atomAddF(base + (h0 + 4) * 8 + g, acc1);
    if (lane == 0) atomAddF(counts + a, 1.0f);
  }
}

// ---------- per-node: mean -> softmax(g) -> att @ vh -> ctx(bf16) ----------
// block = 256 threads, one node per block.
__global__ __launch_bounds__(256) void node_kernel(
    const float* __restrict__ sums, const float* __restrict__ counts,
    const float* __restrict__ vh, unsigned short* __restrict__ ctx, int Nn) {
  __shared__ float sm[64];
  __shared__ float satt[64];
  int n = blockIdx.x;
  int t = threadIdx.x;

  if (t < 64) {
    float c = fmaxf(counts[n], 1.0f);
    sm[t] = sums[(size_t)n * 64 + t] / c;
  }
  __syncthreads();

  if (t < 8) {   // softmax row h = t over 8 g values
    float mx = -1e30f;
    #pragma unroll
    for (int gg = 0; gg < 8; ++gg) mx = fmaxf(mx, sm[t * 8 + gg]);
    float s = 0.f, ex[8];
    #pragma unroll
    for (int gg = 0; gg < 8; ++gg) { ex[gg] = __expf(sm[t * 8 + gg] - mx); s += ex[gg]; }
    float inv = 1.0f / s;
    #pragma unroll
    for (int gg = 0; gg < 8; ++gg) satt[t * 8 + gg] = ex[gg] * inv;
  }
  __syncthreads();

  int h = t >> 5, d = t & 31;
  float acc = 0.f;
  #pragma unroll
  for (int gg = 0; gg < 8; ++gg)
    acc += satt[h * 8 + gg] * vh[(size_t)n * EMB + gg * HDIM + d];
  ctx[(size_t)n * EMB + h * HDIM + d] = f2bf(acc);
}

// ---------- host ----------
extern "C" void kernel_launch(void* const* d_in, const int* in_sizes, int n_in,
                              void* d_out, int out_size, void* d_ws, size_t ws_size,
                              hipStream_t stream) {
  (void)n_in; (void)out_size; (void)ws_size;
  const float* q  = (const float*)d_in[0];
  const float* k  = (const float*)d_in[1];
  const float* v  = (const float*)d_in[2];
  const int*   ei = (const int*)  d_in[3];
  const float* Wq = (const float*)d_in[4];
  const float* bq = (const float*)d_in[5];
  const float* Wk = (const float*)d_in[6];
  const float* bk = (const float*)d_in[7];
  const float* Wv = (const float*)d_in[8];
  const float* bv = (const float*)d_in[9];
  const float* Wc = (const float*)d_in[10];
  const float* bc = (const float*)d_in[11];
  float* out = (float*)d_out;

  const int M = in_sizes[0] / EMB;   // 50000 nodes
  const int E = in_sizes[3] / 2;     // 800000 edges

  char* ws = (char*)d_ws;
  size_t off = 0;
  auto walloc = [&](size_t bytes) -> void* {
    void* p = ws + off;
    off += (bytes + 255) & ~(size_t)255;
    return p;
  };
  unsigned short* q_bf   = (unsigned short*)walloc((size_t)M * EMB * 2);
  unsigned short* k_bf   = (unsigned short*)walloc((size_t)M * EMB * 2);
  unsigned short* v_bf   = (unsigned short*)walloc((size_t)M * EMB * 2);
  unsigned short* Wq_bf  = (unsigned short*)walloc((size_t)EMB * EMB * 2);
  unsigned short* Wk_bf  = (unsigned short*)walloc((size_t)EMB * EMB * 2);
  unsigned short* Wv_bf  = (unsigned short*)walloc((size_t)EMB * EMB * 2);
  unsigned short* Wc_bf  = (unsigned short*)walloc((size_t)EMB * EMB * 2);
  unsigned short* qh_bf  = (unsigned short*)walloc((size_t)M * EMB * 2);
  unsigned short* kh_bf  = (unsigned short*)walloc((size_t)M * EMB * 2);
  float*          vh_f   = (float*)         walloc((size_t)M * EMB * 4);
  float*          sums   = (float*)         walloc((size_t)M * 64 * 4);
  float*          counts = (float*)         walloc((size_t)M * 4);
  unsigned short* ctx_bf = (unsigned short*)walloc((size_t)M * EMB * 2);

  const int nElem = M * EMB;
  const int wElem = EMB * EMB;
  cvt_bf16_kernel<<<(nElem + 255) / 256, 256, 0, stream>>>(q, q_bf, nElem);
  cvt_bf16_kernel<<<(nElem + 255) / 256, 256, 0, stream>>>(k, k_bf, nElem);
  cvt_bf16_kernel<<<(nElem + 255) / 256, 256, 0, stream>>>(v, v_bf, nElem);
  cvt_bf16_kernel<<<(wElem + 255) / 256, 256, 0, stream>>>(Wq, Wq_bf, wElem);
  cvt_bf16_kernel<<<(wElem + 255) / 256, 256, 0, stream>>>(Wk, Wk_bf, wElem);
  cvt_bf16_kernel<<<(wElem + 255) / 256, 256, 0, stream>>>(Wv, Wv_bf, wElem);
  cvt_bf16_kernel<<<(wElem + 255) / 256, 256, 0, stream>>>(Wc, Wc_bf, wElem);

  zero_kernel<<<(M * 64 + 255) / 256, 256, 0, stream>>>(sums, M * 64);
  zero_kernel<<<(M + 255) / 256, 256, 0, stream>>>(counts, M);

  const int gBlocks = (M + 15) / 16;
  gemm_nt_bf16_kernel<<<gBlocks, 128, 0, stream>>>(q_bf, Wq_bf, bq, nullptr, qh_bf, M);
  gemm_nt_bf16_kernel<<<gBlocks, 128, 0, stream>>>(k_bf, Wk_bf, bk, nullptr, kh_bf, M);
  gemm_nt_bf16_kernel<<<gBlocks, 128, 0, stream>>>(v_bf, Wv_bf, bv, vh_f, nullptr, M);

  edge_kernel<<<(E + 7) / 8, 256, 0, stream>>>(qh_bf, kh_bf, ei, sums, counts, E);
  node_kernel<<<M, 256, 0, stream>>>(sums, counts, vh_f, ctx_bf, M);

  gemm_nt_bf16_kernel<<<gBlocks, 128, 0, stream>>>(ctx_bf, Wc_bf, bc, out, nullptr, M);
}
